// RAVENNTM_12103217840276
// MI455X (gfx1250) — matrix operand order repaired and checked by reading, not compiled, not run on hardware
//
#include <hip/hip_runtime.h>
#include <hip/hip_bf16.h>
#include <stdint.h>

// ---------------------------------------------------------------------------
// Types / helpers
// ---------------------------------------------------------------------------
typedef __attribute__((ext_vector_type(16))) __bf16 v16bf;
typedef __attribute__((ext_vector_type(8)))  float  v8f;
typedef __attribute__((ext_vector_type(4)))  int    v4i_;

#if defined(__HIP_DEVICE_COMPILE__) && \
    __has_builtin(__builtin_amdgcn_global_load_async_to_lds_b128) && \
    __has_builtin(__builtin_amdgcn_s_wait_asynccnt)
#define USE_ASYNC 1
#else
#define USE_ASYNC 0
#endif

static __device__ __forceinline__ __bf16 f2bf(float f) {
  unsigned u = __float_as_uint(f);
  unsigned r = u + 0x7fffu + ((u >> 16) & 1u);   // round-to-nearest-even
  unsigned short s = (unsigned short)(r >> 16);
  __bf16 b;
  __builtin_memcpy(&b, &s, 2);
  return b;
}
static __device__ __forceinline__ float bf2f(__bf16 b) {
  unsigned short s;
  __builtin_memcpy(&s, &b, 2);
  return __uint_as_float(((unsigned)s) << 16);
}
static __device__ __forceinline__ float sigm(float x) { return 1.0f / (1.0f + __expf(-x)); }

#if USE_ASYNC
// 32B global -> LDS copy, two async b128 transfers (ASYNCcnt-tracked).
// Builtin signature (from toolchain diagnostics): (AS1 int4*, AS3 int4*, imm
// offset, imm cpol); the instruction offset applies to both addresses.
static __device__ __forceinline__ void async_cp32(const void* g, void* l) {
  __builtin_amdgcn_global_load_async_to_lds_b128(
      (__attribute__((address_space(1))) v4i_*)(g),
      (__attribute__((address_space(3))) v4i_*)(l), 0, 0);
  __builtin_amdgcn_global_load_async_to_lds_b128(
      (__attribute__((address_space(1))) v4i_*)(g),
      (__attribute__((address_space(3))) v4i_*)(l), 16, 0);
}
#endif

// ---------------------------------------------------------------------------
// Weight repack: OIHW fp32 -> W^T [Cout][Kpad] bf16, k = (kh*KW + kw)*Cin + ci,
// K zero-padded to a multiple of 32 -> all conv B reads are 128-bit.
// ---------------------------------------------------------------------------
__global__ void k_pack_w(const float* __restrict__ w, __bf16* __restrict__ wt,
                         int Cout, int Cin, int KH, int KW, int Kp) {
  int K = Cin * KH * KW;
  long total = (long)Cout * Kp;
  for (long i = blockIdx.x * (long)blockDim.x + threadIdx.x; i < total;
       i += (long)gridDim.x * blockDim.x) {
    int co = (int)(i / Kp);
    int k  = (int)(i % Kp);
    float v = 0.0f;
    if (k < K) {
      int ci = k % Cin;
      int rem = k / Cin;
      int kw = rem % KW;
      int kh = rem / KW;
      v = w[(((long)co * Cin + ci) * KH + kh) * KW + kw];
    }
    wt[i] = f2bf(v);
  }
}

// fp32 -> bf16 elementwise (input frames, C=1 so NCHW==NHWC)
__global__ void k_f2bf(const float* __restrict__ in, __bf16* __restrict__ out, long n) {
  for (long i = blockIdx.x * (long)blockDim.x + threadIdx.x; i < n;
       i += (long)gridDim.x * blockDim.x)
    out[i] = f2bf(in[i]);
}

// ---------------------------------------------------------------------------
// Implicit-GEMM conv (NHWC bf16), fused BN affine + optional residual + ReLU.
//   Out[p, co] = sum_k A[p,k] * W[k,co];  p=(n,oh,ow), k=(kh,kw,ci)
// Tile: BM=128 x BN=64 x BK=32; 8 waves; wave w owns rows [w*16,+16) x 64 cols
// -> 1 shared A fragment, 4 accumulators, 4x v_wmma_f32_16x16x32_bf16 / step.
// Fragments are pure ds_load_b128 (A: row-major sA; B: K-contiguous sBt).
// LDS ping-pong; next tile staged via GLOBAL_LOAD_ASYNC_TO_LDS_B128 when the
// toolchain exposes it (s_wait_asynccnt 0 before the barrier), else via
// register staging.  Chunk (kh,kw,ci) state advances incrementally (no divides
// in the K loop).
// ---------------------------------------------------------------------------
#define BM 128
#define BN 64
#define BK 32
#define ASTR 40   // LDS row stride in bf16 (80B: 16B aligned, conflict-free)

__global__ __launch_bounds__(256) void k_conv_wmma(
    const __bf16* __restrict__ in, const __bf16* __restrict__ wt,
    const float* __restrict__ gamma, const float* __restrict__ beta,
    const __bf16* __restrict__ res, __bf16* __restrict__ out,
    int Nimg, int Hin, int Win, int Cin,
    int Hout, int Wout, int Cout,
    int KH, int KW, int stride, int pad, int relu, int K, int Kp) {
  __shared__ __align__(16) __bf16 sA[2][BM][ASTR];
  __shared__ __align__(16) __bf16 sBt[2][BN][ASTR];

  const int tid  = threadIdx.x;
  const int lane = tid & 31;
  const int wv   = tid >> 5;

  const int P   = Nimg * Hout * Wout;
  const int p0  = blockIdx.x * BM;
  const int n0  = blockIdx.y * BN;
  const int HWo = Hout * Wout;
  const bool fastA = (Cin & 31) == 0;   // BK chunk has uniform (kh,kw) -> vector loads

  // This thread's staging slots: A covers (row ar, K-half ah); B covers
  // (col bcol, K-half bh) for tid < 128.
  const int ar = tid >> 1;
  const int ah = tid & 1;
  const int apix = p0 + ar;
  int a_nn = 0, a_oh = 0, a_ow = 0;
  if (apix < P) {
    a_nn = apix / HWo;
    int rest = apix - a_nn * HWo;
    a_oh = rest / Wout;
    a_ow = rest - a_oh * Wout;
  }
  const int bcol = tid >> 1;
  const int bh   = tid & 1;

  // Stage one BK chunk of A into sA[buf].  Fast path: (cib,kw2,kh2) give the
  // chunk's uniform filter tap; per-row validity -> 32B contiguous copy.
  auto stageA = [&](int buf, int k0, int cib, int kw2, int kh2) {
    void* dst = (void*)&sA[buf][ar][ah * 16];
    if (fastA) {
      const __bf16* src = nullptr;
      if (apix < P) {
        int ih = a_oh * stride - pad + kh2;
        int iw = a_ow * stride - pad + kw2;
        if ((unsigned)ih < (unsigned)Hin && (unsigned)iw < (unsigned)Win)
          src = in + ((long)(a_nn * Hin + ih) * Win + iw) * Cin + cib + ah * 16;
      }
#if USE_ASYNC
      if (src) {
        async_cp32(src, dst);
      } else {
        uint4 z = make_uint4(0u, 0u, 0u, 0u);
        ((uint4*)dst)[0] = z;
        ((uint4*)dst)[1] = z;
      }
#else
      uint4 d0 = make_uint4(0u, 0u, 0u, 0u), d1 = d0;
      if (src) {
        d0 = ((const uint4*)src)[0];
        d1 = ((const uint4*)src)[1];
      }
      ((uint4*)dst)[0] = d0;
      ((uint4*)dst)[1] = d1;
#endif
    } else {
      // generic im2col gather (stem conv: Cin==1)
      unsigned short t[16];
      for (int e = 0; e < 16; ++e) {
        int k = k0 + ah * 16 + e;
        float v = 0.0f;
        if (k < K && apix < P) {
          int ci, rem;
          if (Cin == 1) { ci = 0; rem = k; }
          else          { ci = k % Cin; rem = k / Cin; }
          int kw2g = rem % KW, kh2g = rem / KW;
          int ih = a_oh * stride - pad + kh2g;
          int iw = a_ow * stride - pad + kw2g;
          if ((unsigned)ih < (unsigned)Hin && (unsigned)iw < (unsigned)Win)
            v = bf2f(in[((long)(a_nn * Hin + ih) * Win + iw) * Cin + ci]);
        }
        __bf16 bb = f2bf(v);
        __builtin_memcpy(&t[e], &bb, 2);
      }
      __builtin_memcpy((uint4*)dst + 0, &t[0], 16);
      __builtin_memcpy((uint4*)dst + 1, &t[8], 16);
    }
  };
  auto stageB = [&](int buf, int k0) {
    const __bf16* src = wt + (long)(n0 + bcol) * Kp + k0 + bh * 16;
    void* dst = (void*)&sBt[buf][bcol][bh * 16];
#if USE_ASYNC
    async_cp32(src, dst);
#else
    uint4 d0 = ((const uint4*)src)[0];
    uint4 d1 = ((const uint4*)src)[1];
    ((uint4*)dst)[0] = d0;
    ((uint4*)dst)[1] = d1;
#endif
  };

  v8f acc[4];
  acc[0] = v8f{};
  acc[1] = v8f{};
  acc[2] = v8f{};
  acc[3] = v8f{};

  // incremental chunk state (next chunk to stage); valid for the fast path
  int cibN = 0, kwN = 0, khN = 0;
  auto advance = [&]() {
    cibN += BK;
    if (cibN >= Cin) {
      cibN = 0;
      if (++kwN == KW) { kwN = 0; ++khN; }
    }
  };

  // prologue: stage chunk 0 into buffer 0
  stageA(0, 0, 0, 0, 0);
  if (tid < 128) stageB(0, 0);
  advance();
#if USE_ASYNC
  __builtin_amdgcn_s_wait_asynccnt(0);
#endif
  __syncthreads();

  const int nsteps = Kp / BK;
  for (int s = 0; s < nsteps; ++s) {
    const int cur = s & 1;
    // stage next chunk into the other buffer (its last readers finished at the
    // barrier ending step s-1); async copies overlap the WMMAs below.
    if (s + 1 < nsteps) {
      stageA(cur ^ 1, (s + 1) * BK, cibN, kwN, khN);
      if (tid < 128) {
        stageB(cur ^ 1, (s + 1) * BK);
        if (s + 2 < nsteps)
          __builtin_prefetch(wt + (long)(n0 + bcol) * Kp + (s + 2) * BK + bh * 16, 0, 0);
      }
      advance();
    }

    // ---- fragments: pure b128 LDS reads (ISA 7.12.2 layouts) ----
    const int row  = wv * 16 + (lane & 15);
    const int half = lane >> 4;
    v16bf af;
    {
      uint4 lo = *(const uint4*)&sA[cur][row][half * 8];
      uint4 hi = *(const uint4*)&sA[cur][row][16 + half * 8];
      __builtin_memcpy(&af, &lo, 16);
      __builtin_memcpy(reinterpret_cast<char*>(&af) + 16, &hi, 16);
    }
    const int kb = (lane >> 4) * 8;
#pragma unroll
    for (int j = 0; j < 4; ++j) {
      const int col = j * 16 + (lane & 15);
      v16bf bfg;
      uint4 lo = *(const uint4*)&sBt[cur][col][kb];
      uint4 hi = *(const uint4*)&sBt[cur][col][kb + 16];
      __builtin_memcpy(&bfg, &lo, 16);
      __builtin_memcpy(reinterpret_cast<char*>(&bfg) + 16, &hi, 16);
      acc[j] = __builtin_amdgcn_wmma_f32_16x16x32_bf16(false, af, false, bfg,
                                                       (short)0, acc[j], false, false);
    }
#if USE_ASYNC
    __builtin_amdgcn_s_wait_asynccnt(0);
#endif
    __syncthreads();
  }

  // ---- epilogue: BN affine (+res) (+relu), store bf16 NHWC ----
  const int colL   = lane & 15;
  const int rowoff = (lane >> 4) * 8;
#pragma unroll
  for (int j = 0; j < 4; ++j) {
    int co = n0 + j * 16 + colL;
    if (co < Cout) {
      float g = gamma[co], b = beta[co];
#pragma unroll
      for (int r2 = 0; r2 < 8; ++r2) {
        int pix = p0 + wv * 16 + rowoff + r2;
        if (pix < P) {
          float v = acc[j][r2] * g + b;
          long o = (long)pix * Cout + co;
          if (res) v += bf2f(res[o]);
          if (relu) v = v > 0.0f ? v : 0.0f;
          out[o] = f2bf(v);
        }
      }
    }
  }
}

// ---------------------------------------------------------------------------
// 3x3 s2 p1 maxpool, NHWC bf16
// ---------------------------------------------------------------------------
__global__ void k_maxpool(const __bf16* __restrict__ in, __bf16* __restrict__ out,
                          int Nimg, int Hin, int Win, int C, int Hout, int Wout) {
  long total = (long)Nimg * Hout * Wout * C;
  for (long i = blockIdx.x * (long)blockDim.x + threadIdx.x; i < total;
       i += (long)gridDim.x * blockDim.x) {
    int c = (int)(i % C);
    long rest = i / C;
    int ow = (int)(rest % Wout); rest /= Wout;
    int oh = (int)(rest % Hout);
    int n  = (int)(rest / Hout);
    float m = -3.4e38f;
    for (int kh = 0; kh < 3; ++kh) {
      int ih = oh * 2 - 1 + kh;
      if ((unsigned)ih >= (unsigned)Hin) continue;
      for (int kw = 0; kw < 3; ++kw) {
        int iw = ow * 2 - 1 + kw;
        if ((unsigned)iw >= (unsigned)Win) continue;
        float v = bf2f(in[((long)(n * Hin + ih) * Win + iw) * C + c]);
        m = v > m ? v : m;
      }
    }
    out[i] = f2bf(m);
  }
}

// global average pool 7x7, NHWC bf16 -> fp32 feats [Nimg, C]
__global__ void k_avgpool(const __bf16* __restrict__ in, float* __restrict__ feats,
                          int Nimg, int HW, int C) {
  long total = (long)Nimg * C;
  for (long i = blockIdx.x * (long)blockDim.x + threadIdx.x; i < total;
       i += (long)gridDim.x * blockDim.x) {
    int n = (int)(i / C), c = (int)(i % C);
    float s = 0.0f;
    for (int p = 0; p < HW; ++p) s += bf2f(in[((long)n * HW + p) * C + c]);
    feats[i] = s / (float)HW;
  }
}

// ---------------------------------------------------------------------------
// MRA recurrence: single block, 256 threads, fp32.  T=16, B=4, CTRL=128,
// KEY=256, MEM=10, FEAT=512.  M lives in global ws (40KB).
// ---------------------------------------------------------------------------
#define MB 4
#define MCTRL 128
#define MKEY 256
#define MMEM 10
#define MT 16
#define MFEAT 512
#define MEPS 1e-8f

__global__ __launch_bounds__(256) void k_mra(
    const float* __restrict__ feats,   // [64, 512], frame n = b*16 + t
    float* __restrict__ M,             // ws [4][10][256]
    const float* __restrict__ wx, const float* __restrict__ wh,
    const float* __restrict__ lb,
    const float* __restrict__ rkw, const float* __restrict__ rkb,
    const float* __restrict__ wkw, const float* __restrict__ wkb,
    const float* __restrict__ ew,  const float* __restrict__ eb,
    const float* __restrict__ aw,  const float* __restrict__ ab,
    const float* __restrict__ ow,  const float* __restrict__ ob,
    const float* __restrict__ M0,
    float* __restrict__ dout) {
  __shared__ float h[MB][MCTRL], c[MB][MCTRL], r[MB][MKEY];
  __shared__ float gates[MB][4 * MCTRL];
  __shared__ float kv[MB][MKEY], ev[MB][MKEY], av[MB][MKEY];
  __shared__ float dots[MB][MMEM], mnorm[MB][MMEM], knorm[MB], wwv[MB][MMEM];
  __shared__ int   idxs[MB];

  const int tid = threadIdx.x;

  for (int i = tid; i < MB * MCTRL; i += 256) { h[i / MCTRL][i % MCTRL] = 0.0f; c[i / MCTRL][i % MCTRL] = 0.0f; }
  for (int i = tid; i < MB * MKEY;  i += 256) r[i / MKEY][i % MKEY] = 0.0f;
  for (int i = tid; i < MB * MMEM * MKEY; i += 256) M[i] = M0[i % (MMEM * MKEY)];
  __syncthreads();

  for (int t = 0; t < MT; ++t) {
    // gates = [xt, r] @ wx + h @ wh + b
    for (int jj = tid; jj < MB * 4 * MCTRL; jj += 256) {
      int b = jj / (4 * MCTRL), j = jj % (4 * MCTRL);
      const float* xt = feats + (long)(b * MT + t) * MFEAT;
      float s = lb[j];
      for (int k = 0; k < MFEAT; ++k)  s += xt[k]   * wx[(long)k * 512 + j];
      for (int k = 0; k < MKEY; ++k)   s += r[b][k] * wx[(long)(MFEAT + k) * 512 + j];
      for (int k = 0; k < MCTRL; ++k)  s += h[b][k] * wh[(long)k * 512 + j];
      gates[b][j] = s;
    }
    __syncthreads();
    // LSTM cell
    for (int jj = tid; jj < MB * MCTRL; jj += 256) {
      int b = jj / MCTRL, q = jj % MCTRL;
      float gi = gates[b][q], gf = gates[b][MCTRL + q];
      float gg = gates[b][2 * MCTRL + q], go = gates[b][3 * MCTRL + q];
      float cn = sigm(gf) * c[b][q] + sigm(gi) * tanhf(gg);
      c[b][q] = cn;
      h[b][q] = sigm(go) * tanhf(cn);
    }
    __syncthreads();
    // write-head projections: kw, erase, add
    for (int jj = tid; jj < MB * MKEY; jj += 256) {
      int b = jj / MKEY, q = jj % MKEY;
      float sk = wkb[q], se = eb[q], sa = ab[q];
      for (int k = 0; k < MCTRL; ++k) {
        float hv = h[b][k];
        sk += hv * wkw[(long)k * MKEY + q];
        se += hv * ew[(long)k * MKEY + q];
        sa += hv * aw[(long)k * MKEY + q];
      }
      kv[b][q] = sk; ev[b][q] = sigm(se); av[b][q] = tanhf(sa);
    }
    __syncthreads();
    // cosine(kw, M)
    if (tid < MB * MMEM) {
      int b = tid / MMEM, m = tid % MMEM;
      float d = 0.0f, n2 = 0.0f;
      const float* Mr = M + ((long)b * MMEM + m) * MKEY;
      for (int q = 0; q < MKEY; ++q) { d += kv[b][q] * Mr[q]; n2 += Mr[q] * Mr[q]; }
      dots[b][m] = d; mnorm[b][m] = sqrtf(n2);
    }
    if (tid >= 64 && tid < 64 + MB) {
      int b = tid - 64;
      float n2 = 0.0f;
      for (int q = 0; q < MKEY; ++q) n2 += kv[b][q] * kv[b][q];
      knorm[b] = sqrtf(n2);
    }
    __syncthreads();
    // softmax over memory slots
    if (tid < MB) {
      int b = tid;
      float mx = -3.4e38f;
      float cs[MMEM];
      for (int m = 0; m < MMEM; ++m) {
        float v = dots[b][m] / ((knorm[b] + MEPS) * (mnorm[b][m] + MEPS));
        cs[m] = v; mx = v > mx ? v : mx;
      }
      float ssum = 0.0f;
      for (int m = 0; m < MMEM; ++m) { cs[m] = __expf(cs[m] - mx); ssum += cs[m]; }
      for (int m = 0; m < MMEM; ++m) wwv[b][m] = cs[m] / ssum;
    }
    __syncthreads();
    // memory erase/add
    for (int jj = tid; jj < MB * MMEM * MKEY; jj += 256) {
      int b = jj / (MMEM * MKEY);
      int m = (jj / MKEY) % MMEM;
      int q = jj % MKEY;
      float w = wwv[b][m];
      float mv = M[jj];
      M[jj] = mv * (1.0f - w * ev[b][q]) + w * av[b][q];
    }
    __syncthreads();
    // read key (reuse kv)
    for (int jj = tid; jj < MB * MKEY; jj += 256) {
      int b = jj / MKEY, q = jj % MKEY;
      float sk = rkb[q];
      for (int k = 0; k < MCTRL; ++k) sk += h[b][k] * rkw[(long)k * MKEY + q];
      kv[b][q] = sk;
    }
    __syncthreads();
    // cosine(kr, M') + argmax
    if (tid < MB * MMEM) {
      int b = tid / MMEM, m = tid % MMEM;
      float d = 0.0f, n2 = 0.0f;
      const float* Mr = M + ((long)b * MMEM + m) * MKEY;
      for (int q = 0; q < MKEY; ++q) { d += kv[b][q] * Mr[q]; n2 += Mr[q] * Mr[q]; }
      dots[b][m] = d; mnorm[b][m] = sqrtf(n2);
    }
    if (tid >= 64 && tid < 64 + MB) {
      int b = tid - 64;
      float n2 = 0.0f;
      for (int q = 0; q < MKEY; ++q) n2 += kv[b][q] * kv[b][q];
      knorm[b] = sqrtf(n2);
    }
    __syncthreads();
    if (tid < MB) {
      int b = tid;
      float best = -3.4e38f; int bi = 0;
      for (int m = 0; m < MMEM; ++m) {
        float v = dots[b][m] / ((knorm[b] + MEPS) * (mnorm[b][m] + MEPS));
        if (v > best) { best = v; bi = m; }
      }
      idxs[b] = bi;
    }
    __syncthreads();
    for (int jj = tid; jj < MB * MKEY; jj += 256) {
      int b = jj / MKEY, q = jj % MKEY;
      r[b][q] = M[((long)b * MMEM + idxs[b]) * MKEY + q];
    }
    __syncthreads();
    // output (only last step is returned)
    if (t == MT - 1) {
      for (int jj = tid; jj < MB * 38; jj += 256) {
        int b = jj / 38, u = jj % 38;
        float s = ob[u];
        for (int k = 0; k < MCTRL; ++k) s += h[b][k] * ow[(long)k * 38 + u];
        for (int k = 0; k < MKEY;  ++k) s += r[b][k] * ow[(long)(MCTRL + k) * 38 + u];
        if (u < 8)       dout[b * 8 + u] = s;
        else if (u < 17) dout[32 + b * 9 + (u - 8)] = s;
        else             dout[32 + 36 + b * 21 + (u - 17)] = s;
      }
    }
    __syncthreads();
  }
}

// ---------------------------------------------------------------------------
// Host: launch sequence
// ---------------------------------------------------------------------------
struct ConvW { int wi, gi, bi, Cin, Cout, KH, KW; };

static inline size_t al256(size_t x) { return (x + 255) & ~(size_t)255; }
static inline int kpad32(int k) { return (k + 31) & ~31; }

extern "C" void kernel_launch(void* const* d_in, const int* in_sizes, int n_in,
                              void* d_out, int out_size, void* d_ws, size_t ws_size,
                              hipStream_t stream) {
  (void)in_sizes; (void)n_in; (void)out_size; (void)ws_size;
  // Flattened input order: insertion-order recursive walk of setup_inputs():
  // 0:x 1:embedding 2:indicator, then resnet (conv1,bn1.g,bn1.b, layer1..4
  // blocks: conv1,bn1.g,bn1.b,conv2,bn2.g,bn2.b[,down_conv,down_bn.g,down_bn.b]),
  // then mra: lstm.wx,wh,b, rkey.w,b, wkey.w,b, erase.w,b, add.w,b, out.w,b, M0.
  const ConvW cw[20] = {
      {3, 4, 5, 1, 64, 7, 7},          // conv1
      {6, 7, 8, 64, 64, 3, 3},  {9, 10, 11, 64, 64, 3, 3},     // l1b0
      {12, 13, 14, 64, 64, 3, 3}, {15, 16, 17, 64, 64, 3, 3},  // l1b1
      {18, 19, 20, 64, 128, 3, 3}, {21, 22, 23, 128, 128, 3, 3}, {24, 25, 26, 64, 128, 1, 1}, // l2b0
      {27, 28, 29, 128, 128, 3, 3}, {30, 31, 32, 128, 128, 3, 3},                             // l2b1
      {33, 34, 35, 128, 256, 3, 3}, {36, 37, 38, 256, 256, 3, 3}, {39, 40, 41, 128, 256, 1, 1}, // l3b0
      {42, 43, 44, 256, 256, 3, 3}, {45, 46, 47, 256, 256, 3, 3},                               // l3b1
      {48, 49, 50, 256, 512, 3, 3}, {51, 52, 53, 512, 512, 3, 3}, {54, 55, 56, 256, 512, 1, 1}, // l4b0
      {57, 58, 59, 512, 512, 3, 3}, {60, 61, 62, 512, 512, 3, 3},                               // l4b1
  };

  uint8_t* ws = (uint8_t*)d_ws;
  size_t off = 0;
  auto alloc = [&](size_t bytes) { size_t o = off; off = al256(off + bytes); return o; };

  // packed (transposed, K-padded) weights
  size_t wtoff[20];
  int kp[20];
  for (int i = 0; i < 20; ++i) {
    kp[i] = kpad32(cw[i].Cin * cw[i].KH * cw[i].KW);
    wtoff[i] = alloc((size_t)cw[i].Cout * kp[i] * sizeof(__bf16));
  }
  const long NX = 4L * 16 * 224 * 224;
  size_t xbf_o  = alloc((size_t)NX * 2);
  size_t A1_o   = alloc((size_t)64 * 112 * 112 * 64 * 2);  // conv1 out; recycled as T0..T3
  size_t P_o    = alloc((size_t)64 * 56 * 56 * 64 * 2);    // maxpool out
  size_t feat_o = alloc((size_t)64 * 512 * 4);
  size_t M_o    = alloc((size_t)4 * 10 * 256 * 4);

  __bf16* xbf = (__bf16*)(ws + xbf_o);
  __bf16* A1  = (__bf16*)(ws + A1_o);
  __bf16* Pb  = (__bf16*)(ws + P_o);
  float*  fea = (float*)(ws + feat_o);
  float*  Mws = (float*)(ws + M_o);
  const size_t Tsz = (size_t)64 * 56 * 56 * 64;  // bf16 elements, largest block buf
  __bf16* T0 = A1;
  __bf16* T1 = A1 + Tsz;
  __bf16* T2 = A1 + 2 * Tsz;
  __bf16* T3 = A1 + 3 * Tsz;

  // 1) pack all conv weights
  for (int i = 0; i < 20; ++i) {
    long total = (long)cw[i].Cout * kp[i];
    int blocks = (int)((total + 255) / 256);
    k_pack_w<<<blocks, 256, 0, stream>>>((const float*)d_in[cw[i].wi],
                                         (__bf16*)(ws + wtoff[i]),
                                         cw[i].Cout, cw[i].Cin, cw[i].KH, cw[i].KW, kp[i]);
  }
  // 2) input -> bf16
  k_f2bf<<<(int)((NX + 255) / 256), 256, 0, stream>>>((const float*)d_in[0], xbf, NX);

  auto conv = [&](int ci, const __bf16* in, const __bf16* res, __bf16* out,
                  int Hin, int Win, int Hout, int Wout, int s, int p, int relu) {
    int P = 64 * Hout * Wout;
    dim3 grid((P + BM - 1) / BM, cw[ci].Cout / BN);
    k_conv_wmma<<<grid, 256, 0, stream>>>(
        in, (const __bf16*)(ws + wtoff[ci]),
        (const float*)d_in[cw[ci].gi], (const float*)d_in[cw[ci].bi],
        res, out, 64, Hin, Win, cw[ci].Cin, Hout, Wout, cw[ci].Cout,
        cw[ci].KH, cw[ci].KW, s, p, relu,
        cw[ci].Cin * cw[ci].KH * cw[ci].KW, kp[ci]);
  };

  // 3) stem
  conv(0, xbf, nullptr, A1, 224, 224, 112, 112, 2, 3, 1);
  {
    long total = 64L * 56 * 56 * 64;
    k_maxpool<<<(int)((total + 255) / 256), 256, 0, stream>>>(A1, Pb, 64, 112, 112, 64, 56, 56);
  }
  // 4) residual layers (y=conv1 relu; [ds]; out=conv2+res relu)
  // layer1
  conv(1, Pb, nullptr, T0, 56, 56, 56, 56, 1, 1, 1);
  conv(2, T0, Pb, T1, 56, 56, 56, 56, 1, 1, 1);
  conv(3, T1, nullptr, T0, 56, 56, 56, 56, 1, 1, 1);
  conv(4, T0, T1, T2, 56, 56, 56, 56, 1, 1, 1);
  // layer2
  conv(5, T2, nullptr, T0, 56, 56, 28, 28, 2, 1, 1);
  conv(7, T2, nullptr, T1, 56, 56, 28, 28, 2, 0, 0);   // downsample 1x1
  conv(6, T0, T1, T3, 28, 28, 28, 28, 1, 1, 1);
  conv(8, T3, nullptr, T0, 28, 28, 28, 28, 1, 1, 1);
  conv(9, T0, T3, T1, 28, 28, 28, 28, 1, 1, 1);
  // layer3
  conv(10, T1, nullptr, T0, 28, 28, 14, 14, 2, 1, 1);
  conv(12, T1, nullptr, T2, 28, 28, 14, 14, 2, 0, 0);
  conv(11, T0, T2, T3, 14, 14, 14, 14, 1, 1, 1);
  conv(13, T3, nullptr, T0, 14, 14, 14, 14, 1, 1, 1);
  conv(14, T0, T3, T1, 14, 14, 14, 14, 1, 1, 1);
  // layer4
  conv(15, T1, nullptr, T0, 14, 14, 7, 7, 2, 1, 1);
  conv(17, T1, nullptr, T2, 14, 14, 7, 7, 2, 0, 0);
  conv(16, T0, T2, T3, 7, 7, 7, 7, 1, 1, 1);
  conv(18, T3, nullptr, T0, 7, 7, 7, 7, 1, 1, 1);
  conv(19, T0, T3, T1, 7, 7, 7, 7, 1, 1, 1);
  // 5) global average pool -> feats [64,512]
  k_avgpool<<<128, 256, 0, stream>>>(T1, fea, 64, 49, 512);
  // 6) MRA recurrence -> d_out (152 floats: [4,8]+[4,9]+[4,21])
  k_mra<<<1, 256, 0, stream>>>(
      fea, Mws,
      (const float*)d_in[63], (const float*)d_in[64], (const float*)d_in[65],
      (const float*)d_in[66], (const float*)d_in[67],
      (const float*)d_in[68], (const float*)d_in[69],
      (const float*)d_in[70], (const float*)d_in[71],
      (const float*)d_in[72], (const float*)d_in[73],
      (const float*)d_in[74], (const float*)d_in[75],
      (const float*)d_in[76], (float*)d_out);
}
